// down_5085241279031
// MI455X (gfx1250) — matrix operand order repaired and checked by reading, not compiled, not run on hardware
//
#include <hip/hip_runtime.h>

// ---------------------------------------------------------------- constants
#define NVI 524288   // 2*64^3 input voxels
#define NVD 65536    // 2*32^3 downsampled voxels
#define CU  64
#define CD  128
#define BN_EPS 1e-4f

// ------------------------------------------------- async-to-LDS capability
#if defined(__gfx1250__) && \
    __has_builtin(__builtin_amdgcn_global_load_async_to_lds_b128) && \
    __has_builtin(__builtin_amdgcn_s_wait_asynccnt)
#define ASYNC_OK 1
#else
#define ASYNC_OK 0
#endif

typedef __attribute__((ext_vector_type(4))) int v4i;

#if ASYNC_OK
typedef __attribute__((address_space(1))) v4i* gptr_v4i;  // global v4i*
typedef __attribute__((address_space(3))) v4i* lptr_v4i;  // LDS v4i*

__device__ __forceinline__ void cp_async16(const void* g, void* l) {
  // per-lane 16B global -> LDS, tracked by ASYNCcnt (no VGPR round trip)
  __builtin_amdgcn_global_load_async_to_lds_b128(
      (gptr_v4i)(unsigned long long)g,
      (lptr_v4i)(unsigned int)(unsigned long long)l,
      0, 0);
}
#endif

// ---------------------------------------------------------------- WMMA types
typedef __attribute__((ext_vector_type(16))) __bf16 v16bf;
typedef __attribute__((ext_vector_type(8)))  float  v8f;

union Frag16 { v16bf v; unsigned int u[8]; };

__device__ __forceinline__ v8f zero_v8f() {
  v8f z;
#pragma unroll
  for (int i = 0; i < 8; ++i) z[i] = 0.0f;
  return z;
}

__device__ __forceinline__ v8f wmma_bf16(const Frag16& a, const Frag16& b, v8f c) {
  return __builtin_amdgcn_wmma_f32_16x16x32_bf16(
      false, a.v, false, b.v, (short)0, c, false, false);
}

__device__ __forceinline__ unsigned short f2bf(float f) {
  unsigned int u = __float_as_uint(f);
  unsigned int r = u + 0x7FFFu + ((u >> 16) & 1u);  // RNE
  return (unsigned short)(r >> 16);
}

// One K-chunk of math: B frag shared per wave, 4 M-subtiles accumulate.
__device__ __forceinline__ void tile_mma(const unsigned char* sAp, const unsigned char* sBp,
                                         int wv, int lane, v8f acc[4]) {
  const int lh = lane >> 4, lr = lane & 15;
  Frag16 bf;  // B: lane = N col (lane&15), K pairs along VGPRs with half offset
  const unsigned short* bp = (const unsigned short*)sBp + wv * 16 + lr;
#pragma unroll
  for (int v = 0; v < 8; ++v) {
    const int k = lh * 8 + (v < 4 ? 2 * v : 16 + 2 * (v - 4));
    bf.u[v] = (unsigned int)bp[k * 128] | ((unsigned int)bp[(k + 1) * 128] << 16);
  }
#pragma unroll
  for (int mt = 0; mt < 4; ++mt) {
    Frag16 af;  // A: lane = M row (lane&15), K pairs along VGPRs with half offset
    const unsigned int* ap = (const unsigned int*)(sAp + (mt * 16 + lr) * 64);
    const int kb = lh * 8;
#pragma unroll
    for (int v = 0; v < 8; ++v) {
      const int k = kb + (v < 4 ? 2 * v : 16 + 2 * (v - 4));
      af.u[v] = ap[k >> 1];
    }
    acc[mt] = wmma_bf16(af, bf, acc[mt]);
  }
}

// ---------------------------------------------------------------- mask prep
__global__ __launch_bounds__(256)
void mask_prep(const unsigned char* __restrict__ mask, float* __restrict__ mdf,
               float* __restrict__ counts) {
  __shared__ float s0[256], s1[256];
  const int tid = threadIdx.x;
  float cin = 0.f, cdn = 0.f;
  for (int v = blockIdx.x * 256 + tid; v < NVD; v += gridDim.x * 256) {
    const int b = v >> 15, rem = v & 32767;
    const int z = rem >> 10, y = (rem >> 5) & 31, x = rem & 31;
    int cnt = 0;
#pragma unroll
    for (int kz = 0; kz < 2; ++kz)
#pragma unroll
      for (int ky = 0; ky < 2; ++ky)
#pragma unroll
        for (int kx = 0; kx < 2; ++kx)
          cnt += mask[(((b * 64 + 2 * z + kz) * 64) + (2 * y + ky)) * 64 + (2 * x + kx)] ? 1 : 0;
    mdf[v] = cnt ? 1.0f : 0.0f;
    cin += (float)cnt;
    cdn += cnt ? 1.0f : 0.0f;
  }
  s0[tid] = cin; s1[tid] = cdn;
  __syncthreads();
  for (int off = 128; off > 0; off >>= 1) {
    if (tid < off) { s0[tid] += s0[tid + off]; s1[tid] += s1[tid + off]; }
    __syncthreads();
  }
  if (tid == 0) { atomicAdd(&counts[0], s0[0]); atomicAdd(&counts[1], s1[0]); }
}

// ---------------------------------------------------------------- BN stats
__global__ __launch_bounds__(256)
void stats_pass(const float* __restrict__ x, int nvox, int cmask, int cshift,
                float* __restrict__ sum, float* __restrict__ sq) {
  __shared__ float ss[256], sqs[256];
  const int tid = threadIdx.x;
  const int c = tid & cmask;
  const int g = tid >> cshift;
  const int V = 256 >> cshift;
  const int C = cmask + 1;
  float s = 0.f, q = 0.f;
  for (long v = (long)blockIdx.x * V + g; v < nvox; v += (long)gridDim.x * V) {
    const float t = x[v * C + c];
    s += t; q += t * t;
  }
  ss[tid] = s; sqs[tid] = q;
  __syncthreads();
  if (g == 0) {
    for (int i = 1; i < V; ++i) { s += ss[i * C + c]; q += sqs[i * C + c]; }
    atomicAdd(&sum[c], s);
    atomicAdd(&sq[c], q);
  }
}

__global__ void finalize_stats(const float* __restrict__ sum, const float* __restrict__ sq,
                               const float* __restrict__ ncnt,
                               const float* __restrict__ g, const float* __restrict__ b,
                               float* __restrict__ scale, float* __restrict__ shift) {
  const int c = threadIdx.x;
  const float n = ncnt[0];
  const float mean = sum[c] / n;
  const float var = sq[c] / n - mean * mean;
  const float inv = rsqrtf(var + BN_EPS);
  scale[c] = g[c] * inv;
  shift[c] = b[c] - mean * g[c] * inv;
}

// -------------------------------------------------- BN+ReLU+mask -> bf16 A
__global__ __launch_bounds__(256)
void bnrelu_bf16(const float* __restrict__ x, const float* __restrict__ scale,
                 const float* __restrict__ shift, const unsigned char* __restrict__ mask_b,
                 const float* __restrict__ mask_f, int cmask, int cshift,
                 unsigned short* __restrict__ dst, long total) {
  for (long i = (long)blockIdx.x * blockDim.x + threadIdx.x; i < total;
       i += (long)gridDim.x * blockDim.x) {
    const int c = (int)(i & (long)cmask);
    const long v = i >> cshift;
    const float m = mask_b ? (mask_b[v] ? 1.0f : 0.0f) : mask_f[v];
    const float y = fmaxf(fmaf(x[i], scale[c], shift[c]), 0.0f) * m;
    dst[i] = f2bf(y);
  }
}

__global__ __launch_bounds__(256)
void f32_to_bf16(const float* __restrict__ src, unsigned short* __restrict__ dst, int n) {
  for (int i = blockIdx.x * blockDim.x + threadIdx.x; i < n; i += gridDim.x * blockDim.x)
    dst[i] = f2bf(src[i]);
}

// ---------------------------------------------------------------- conv k=2 s=2
// Implicit GEMM: M-tile=64 voxels, N=128, K=512 in 16 chunks of 32.
// Async path: 2-deep LDS pipeline; loads of chunk k+1 overlap WMMA of chunk k.
__global__ __launch_bounds__(256)
void conv_down_wmma(const unsigned short* __restrict__ abf,
                    const unsigned short* __restrict__ wbf,
                    const float* __restrict__ mdf, float* __restrict__ out) {
  const int tid = threadIdx.x;
  const int wv = tid >> 5, lane = tid & 31;
  const int lh = lane >> 4, lr = lane & 15;
  const int vbase = blockIdx.x * 64;

  const int arow = tid >> 2, apart = tid & 3;   // A staging role
  const int av = vbase + arow;
  const int b = av >> 15, rem = av & 32767;
  const int z = rem >> 10, y = (rem >> 5) & 31, x = rem & 31;

  v8f acc[4];
#pragma unroll
  for (int i = 0; i < 4; ++i) acc[i] = zero_v8f();

#if ASYNC_OK
  __shared__ __align__(16) unsigned char sA[2][4096];  // 64 rows x 32 bf16
  __shared__ __align__(16) unsigned char sB[2][8192];  // 32 k x 128 n bf16
  auto stage = [&](int kc, int buf) {
    const int nb = kc >> 1, cc = kc & 1;
    const int kz = nb >> 2, ky = (nb >> 1) & 1, kx = nb & 1;
    const long ivox = (((long)(b * 64 + 2 * z + kz) * 64 + (2 * y + ky)) * 64 + (2 * x + kx));
    cp_async16((const char*)abf + ivox * 128 + cc * 64 + apart * 16,
               &sA[buf][arow * 64 + apart * 16]);
    const char* wsrc = (const char*)wbf + (size_t)kc * 8192 + tid * 16;
    cp_async16(wsrc, &sB[buf][tid * 16]);
    cp_async16(wsrc + 4096, &sB[buf][tid * 16 + 4096]);
  };
  stage(0, 0);
  for (int kc = 0; kc < 15; ++kc) {
    stage(kc + 1, (kc + 1) & 1);                  // prefetch next chunk
    __builtin_amdgcn_s_wait_asynccnt(3);          // chunk kc's 3 ops complete
    __syncthreads();
    tile_mma(sA[kc & 1], sB[kc & 1], wv, lane, acc);
    __syncthreads();                              // protect buffer before reuse
  }
  __builtin_amdgcn_s_wait_asynccnt(0);
  __syncthreads();
  tile_mma(sA[1], sB[1], wv, lane, acc);
#else
  __shared__ __align__(16) unsigned char sA1[4096];
  __shared__ __align__(16) unsigned char sB1[8192];
  for (int kc = 0; kc < 16; ++kc) {
    const int nb = kc >> 1, cc = kc & 1;
    const int kz = nb >> 2, ky = (nb >> 1) & 1, kx = nb & 1;
    const long ivox = (((long)(b * 64 + 2 * z + kz) * 64 + (2 * y + ky)) * 64 + (2 * x + kx));
    const uint4 va = *(const uint4*)((const char*)abf + ivox * 128 + cc * 64 + apart * 16);
    *(uint4*)(sA1 + arow * 64 + apart * 16) = va;
    const uint4* wsrc = (const uint4*)((const char*)wbf + (size_t)kc * 8192);
    ((uint4*)sB1)[tid] = wsrc[tid];
    ((uint4*)sB1)[tid + 256] = wsrc[tid + 256];
    __syncthreads();
    tile_mma(sA1, sB1, wv, lane, acc);
    __syncthreads();
  }
#endif

#pragma unroll
  for (int mt = 0; mt < 4; ++mt)
#pragma unroll
    for (int j = 0; j < 8; ++j) {
      const int row = mt * 16 + lh * 8 + j;       // C/D: M = j + 8*lane_half
      const int v = vbase + row;
      out[(size_t)v * 128 + wv * 16 + lr] = acc[mt][j] * mdf[v];
    }
}

// ---------------------------------------------------------------- conv 3x3x3 SAME
// K = 27*128 = 3456 in 108 chunks of 32; optional fused residual add.
// Async staging can't predicate the zero halo (masked lanes skip the LDS
// write), so OOB lanes stream from a zero page in global memory instead.
__global__ __launch_bounds__(256)
void conv3_wmma(const unsigned short* __restrict__ abf,
                const unsigned short* __restrict__ wbf,
                const float* __restrict__ mdf, const float* __restrict__ addsrc,
                const unsigned char* __restrict__ zpage, float* __restrict__ out) {
  const int tid = threadIdx.x;
  const int wv = tid >> 5, lane = tid & 31;
  const int lh = lane >> 4, lr = lane & 15;
  const int vbase = blockIdx.x * 64;

  const int arow = tid >> 2, apart = tid & 3;
  const int av = vbase + arow;
  const int b = av >> 15, rem = av & 32767;
  const int z = rem >> 10, y = (rem >> 5) & 31, x = rem & 31;

  v8f acc[4];
#pragma unroll
  for (int i = 0; i < 4; ++i) acc[i] = zero_v8f();

#if ASYNC_OK
  __shared__ __align__(16) unsigned char sA[2][4096];
  __shared__ __align__(16) unsigned char sB[2][8192];
  auto stage = [&](int kc, int buf) {
    const int nb = kc >> 2, cc = kc & 3;
    const int dz = nb / 9 - 1, dy = (nb / 3) % 3 - 1, dx = nb % 3 - 1;
    const int iz = z + dz, iy = y + dy, ix = x + dx;
    const bool ok = ((unsigned)iz < 32u) && ((unsigned)iy < 32u) && ((unsigned)ix < 32u);
    const long ivox = (((long)(b * 32 + iz) * 32 + iy) * 32 + ix);
    const char* gA = ok ? (const char*)abf + ivox * 256 + cc * 64 + apart * 16
                        : (const char*)zpage + apart * 16;   // zero halo
    cp_async16(gA, &sA[buf][arow * 64 + apart * 16]);
    const char* wsrc = (const char*)wbf + (size_t)kc * 8192 + tid * 16;
    cp_async16(wsrc, &sB[buf][tid * 16]);
    cp_async16(wsrc + 4096, &sB[buf][tid * 16 + 4096]);
  };
  stage(0, 0);
  for (int kc = 0; kc < 107; ++kc) {
    stage(kc + 1, (kc + 1) & 1);
    __builtin_amdgcn_s_wait_asynccnt(3);
    __syncthreads();
    tile_mma(sA[kc & 1], sB[kc & 1], wv, lane, acc);
    __syncthreads();
  }
  __builtin_amdgcn_s_wait_asynccnt(0);
  __syncthreads();
  tile_mma(sA[1], sB[1], wv, lane, acc);
#else
  __shared__ __align__(16) unsigned char sA1[4096];
  __shared__ __align__(16) unsigned char sB1[8192];
  (void)zpage;
  for (int kc = 0; kc < 108; ++kc) {
    const int nb = kc >> 2, cc = kc & 3;
    const int dz = nb / 9 - 1, dy = (nb / 3) % 3 - 1, dx = nb % 3 - 1;
    const int iz = z + dz, iy = y + dy, ix = x + dx;
    const bool ok = ((unsigned)iz < 32u) && ((unsigned)iy < 32u) && ((unsigned)ix < 32u);
    uint4 va = make_uint4(0u, 0u, 0u, 0u);
    if (ok) {
      const long ivox = (((long)(b * 32 + iz) * 32 + iy) * 32 + ix);
      va = *(const uint4*)((const char*)abf + ivox * 256 + cc * 64 + apart * 16);
    }
    *(uint4*)(sA1 + arow * 64 + apart * 16) = va;
    const uint4* wsrc = (const uint4*)((const char*)wbf + (size_t)kc * 8192);
    ((uint4*)sB1)[tid] = wsrc[tid];
    ((uint4*)sB1)[tid + 256] = wsrc[tid + 256];
    __syncthreads();
    tile_mma(sA1, sB1, wv, lane, acc);
    __syncthreads();
  }
#endif

#pragma unroll
  for (int mt = 0; mt < 4; ++mt)
#pragma unroll
    for (int j = 0; j < 8; ++j) {
      const int row = mt * 16 + lh * 8 + j;
      const int v = vbase + row;
      const size_t idx = (size_t)v * 128 + wv * 16 + lr;
      const float base = addsrc ? addsrc[idx] : 0.0f;
      out[idx] = acc[mt][j] * mdf[v] + base;
    }
}

// ---------------------------------------------------------------- launcher
extern "C" void kernel_launch(void* const* d_in, const int* in_sizes, int n_in,
                              void* d_out, int out_size, void* d_ws, size_t ws_size,
                              hipStream_t stream) {
  (void)in_sizes; (void)n_in; (void)out_size; (void)ws_size;
  const float* x            = (const float*)d_in[0];
  const unsigned char* mask = (const unsigned char*)d_in[1];
  const float* bn0_g        = (const float*)d_in[2];
  const float* bn0_b        = (const float*)d_in[3];
  const float* w_down       = (const float*)d_in[4];
  const float* rg1[2] = {(const float*)d_in[5],  (const float*)d_in[11]};
  const float* rb1[2] = {(const float*)d_in[6],  (const float*)d_in[12]};
  const float* rw1[2] = {(const float*)d_in[7],  (const float*)d_in[13]};
  const float* rg2[2] = {(const float*)d_in[8],  (const float*)d_in[14]};
  const float* rb2[2] = {(const float*)d_in[9],  (const float*)d_in[15]};
  const float* rw2[2] = {(const float*)d_in[10], (const float*)d_in[16]};
  float* out = (float*)d_out;

  // workspace layout (bytes)
  char* ws = (char*)d_ws;
  unsigned short* abf = (unsigned short*)(ws);              // 67,108,864 B bf16 acts
  float* tbuf = (float*)(ws + 67108864);                    // 33,554,432 B branch temp
  float* mdf  = (float*)(ws + 100663296);                   //    262,144 B dn mask
  unsigned short* wbf = (unsigned short*)(ws + 100925440);  //    884,736 B bf16 weights
  float* stats = (float*)(ws + 101810176);                  //      4,096 B
  unsigned char* zpage = (unsigned char*)(ws + 101814272);  //        256 B zero halo page
  float* ssum = stats, *ssq = stats + 128, *sscale = stats + 256,
       * sshift = stats + 384, *scnt = stats + 512;

  (void)hipMemsetAsync(stats, 0, 4096 + 256, stream);       // stats + zero page
  mask_prep<<<256, 256, 0, stream>>>(mask, mdf, scnt);

  // ---- BN0(Cu) + ReLU + mask -> bf16, then downsample conv
  stats_pass<<<256, 256, 0, stream>>>(x, NVI, CU - 1, 6, ssum, ssq);
  finalize_stats<<<1, CU, 0, stream>>>(ssum, ssq, scnt + 0, bn0_g, bn0_b, sscale, sshift);
  bnrelu_bf16<<<4096, 256, 0, stream>>>(x, sscale, sshift, mask, nullptr, CU - 1, 6,
                                        abf, (long)NVI * CU);
  f32_to_bf16<<<256, 256, 0, stream>>>(w_down, wbf, 512 * 128);
  conv_down_wmma<<<NVD / 64, 256, 0, stream>>>(abf, wbf, mdf, out);

  // ---- residual blocks: h += SMConv(BNReLU(SMConv(BNReLU(h))))
  for (int r = 0; r < 2; ++r) {
    (void)hipMemsetAsync(ssum, 0, 1024, stream);
    stats_pass<<<256, 256, 0, stream>>>(out, NVD, CD - 1, 7, ssum, ssq);
    finalize_stats<<<1, CD, 0, stream>>>(ssum, ssq, scnt + 1, rg1[r], rb1[r], sscale, sshift);
    bnrelu_bf16<<<4096, 256, 0, stream>>>(out, sscale, sshift, nullptr, mdf, CD - 1, 7,
                                          abf, (long)NVD * CD);
    f32_to_bf16<<<256, 256, 0, stream>>>(rw1[r], wbf, 3456 * 128);
    conv3_wmma<<<NVD / 64, 256, 0, stream>>>(abf, wbf, mdf, nullptr, zpage, tbuf);

    (void)hipMemsetAsync(ssum, 0, 1024, stream);
    stats_pass<<<256, 256, 0, stream>>>(tbuf, NVD, CD - 1, 7, ssum, ssq);
    finalize_stats<<<1, CD, 0, stream>>>(ssum, ssq, scnt + 1, rg2[r], rb2[r], sscale, sshift);
    bnrelu_bf16<<<4096, 256, 0, stream>>>(tbuf, sscale, sshift, nullptr, mdf, CD - 1, 7,
                                          abf, (long)NVD * CD);
    f32_to_bf16<<<256, 256, 0, stream>>>(rw2[r], wbf, 3456 * 128);
    conv3_wmma<<<NVD / 64, 256, 0, stream>>>(abf, wbf, mdf, out, zpage, out);
  }
}
